// GlobalNoPEBlock_41764261986322
// MI455X (gfx1250) — matrix operand order repaired and checked by reading, not compiled, not run on hardware
//
#include <hip/hip_runtime.h>
#include <hip/hip_bf16.h>
#include <math.h>

// ---------------- types / constants ----------------
typedef unsigned short u16t;
typedef __attribute__((ext_vector_type(16))) __bf16 v16bf;
typedef __attribute__((ext_vector_type(8)))  float  v8f;

#define D_MODEL   1024
#define N_HEADS   16
#define DK        64
#define DV        64
#define D_FF      4096
#define GWTOK     4
#define GROUP_KV  4
#define SEQ       2048
#define BATCH     2
#define T_REAL    (SEQ + GWTOK)   // 2052
#define TP        2064            // t padded to /16
#define TPAD      2096            // Vt row length (TP + 32 key-chunk slack), /16
#define MQ        (BATCH * TP)    // 4128 rows for q/k/v gemms
#define MY        (BATCH * SEQ)   // 4096 rows for y/ffn gemms
#define KVW       (GROUP_KV * DK) // 256
#define QK_SCALE  0.125f          // 1/sqrt(64)
#define EPSV      1e-6f

// ---------------- bf16 helpers ----------------
__device__ __forceinline__ u16t f2bf(float f) {
  union { float f; unsigned u; } c; c.f = f;
  unsigned u = c.u;
  u += 0x7fffu + ((u >> 16) & 1u);        // round-to-nearest-even
  return (u16t)(u >> 16);
}

union FragBF { v16bf v; u16t s[16]; uint4 q[2]; };

__device__ __forceinline__ v8f wmma_bf16(v16bf a, v16bf b, v8f c) {
  return __builtin_amdgcn_wmma_f32_16x16x32_bf16(false, a, false, b, (short)0, c,
                                                 false, false);
}

// A-frag (16x32, bf16), A row-major [.. x lda]. Contiguous K layout (matches B):
// lanes 0-15: row M=lane, K 0-15 ; lanes 16-31: row M=lane-16, K 16-31
__device__ __forceinline__ v16bf load_a_rm(const u16t* A, int lda, int r0, int k0) {
  int l = threadIdx.x & 31;
  int half = l >> 4;
  int m = l & 15;
  const u16t* p = A + (size_t)(r0 + m) * lda + k0 + half * 16;
  FragBF a;
  a.q[0] = *(const uint4*)(p);        // K +0..7   (or +16..23)
  a.q[1] = *(const uint4*)(p + 8);    // K +8..15  (or +24..31)
  return a.v;
}

// B-frag (32x16, bf16) where logical B[k][n] = T[n][k], T row-major with ldt.
// Contiguous-K B layout used here: VGPR j, lanes 0-15: K=2j,2j+1 at N=lane ; lanes 16-31: K=16+2j,..
// -> per lane: 16 contiguous K values of one column = two b128 loads.
__device__ __forceinline__ v16bf load_b_tr(const u16t* T, int ldt, int k0, int n0) {
  int l = threadIdx.x & 31;
  int half = l >> 4;
  const u16t* p = T + (size_t)(n0 + (l & 15)) * ldt + k0 + half * 16;
  FragBF b;
  b.q[0] = *(const uint4*)(p);
  b.q[1] = *(const uint4*)(p + 8);
  return b.v;
}

// ---------------- kernels ----------------
// cast fp32 [K][N] row-major -> bf16 transposed [N][K] row-major.
// 32x32 LDS-tiled: coalesced reads AND writes (K, N multiples of 32).
__global__ __launch_bounds__(256)
void cast_transpose_kernel(const float* __restrict__ in, u16t* __restrict__ out,
                           int K, int N) {
  __shared__ u16t tile[32][34];          // +2 u16 pad: avoid bank conflicts
  int n0 = blockIdx.x * 32;
  int k0 = blockIdx.y * 32;
  int tx = threadIdx.x & 31;
  int ty = threadIdx.x >> 5;             // 0..7
#pragma unroll
  for (int r = ty; r < 32; r += 8)
    tile[r][tx] = f2bf(in[(size_t)(k0 + r) * N + n0 + tx]);
  __syncthreads();
#pragma unroll
  for (int r = ty; r < 32; r += 8)
    out[(size_t)(n0 + r) * K + k0 + tx] = tile[tx][r];
}

// v [BATCH][TP][KVW] bf16 -> vt [BATCH][KVW][TPAD] bf16 (zero-padded tail),
// 32x32 LDS-tiled transpose with token-edge guards.
__global__ __launch_bounds__(256)
void transpose_v_kernel(const u16t* __restrict__ v, u16t* __restrict__ vt) {
  __shared__ u16t tile[32][34];
  int b    = blockIdx.z;
  int tok0 = blockIdx.x * 32;
  int col0 = blockIdx.y * 32;
  int tx = threadIdx.x & 31;
  int ty = threadIdx.x >> 5;
#pragma unroll
  for (int r = ty; r < 32; r += 8) {
    int tok = tok0 + r;
    u16t val = 0;
    if (tok < TP) val = v[((size_t)b * TP + tok) * KVW + col0 + tx];
    tile[r][tx] = val;
  }
  __syncthreads();
#pragma unroll
  for (int r = ty; r < 32; r += 8) {
    int tok = tok0 + tx;
    if (tok < TPAD)
      vt[((size_t)b * KVW + col0 + r) * TPAD + tok] = tile[tx][r];
  }
}

// h[b][row][:] : row<GW -> gw token (raw), GW<=row<T_REAL -> rms(x), else zero. bf16 out.
__global__ __launch_bounds__(256)
void prep_h_kernel(const float* __restrict__ x, const float* __restrict__ nw,
                   const float* __restrict__ gw, u16t* __restrict__ h) {
  int row = blockIdx.x;
  int b   = blockIdx.y;
  int tid = threadIdx.x;
  u16t* out = h + ((size_t)b * TP + row) * D_MODEL;
  if (row < GWTOK) {
#pragma unroll
    for (int j = 0; j < 4; ++j) out[tid + j*256] = f2bf(gw[row * D_MODEL + tid + j*256]);
    return;
  }
  if (row >= T_REAL) {
#pragma unroll
    for (int j = 0; j < 4; ++j) out[tid + j*256] = 0;
    return;
  }
  const float* xr = x + ((size_t)b * SEQ + (row - GWTOK)) * D_MODEL;
  float vals[4]; float ss = 0.f;
#pragma unroll
  for (int j = 0; j < 4; ++j) { vals[j] = xr[tid + j*256]; ss += vals[j]*vals[j]; }
  for (int m = 16; m >= 1; m >>= 1) ss += __shfl_xor(ss, m, 32);
  __shared__ float red[8];
  if ((tid & 31) == 0) red[tid >> 5] = ss;
  __syncthreads();
  float tot = 0.f;
#pragma unroll
  for (int w = 0; w < 8; ++w) tot += red[w];
  float rr = rsqrtf(tot / (float)D_MODEL + EPSV);
#pragma unroll
  for (int j = 0; j < 4; ++j)
    out[tid + j*256] = f2bf(vals[j] * rr * nw[tid + j*256]);
}

// z[row][:] = rms(y[row][:]) * w  (f32 in, bf16 out), D_MODEL cols
__global__ __launch_bounds__(256)
void rms_rows_kernel(const float* __restrict__ y, const float* __restrict__ w,
                     u16t* __restrict__ z) {
  size_t row = blockIdx.x;
  int tid = threadIdx.x;
  const float* yr = y + row * D_MODEL;
  u16t* zr = z + row * D_MODEL;
  float vals[4]; float ss = 0.f;
#pragma unroll
  for (int j = 0; j < 4; ++j) { vals[j] = yr[tid + j*256]; ss += vals[j]*vals[j]; }
  for (int m = 16; m >= 1; m >>= 1) ss += __shfl_xor(ss, m, 32);
  __shared__ float red[8];
  if ((tid & 31) == 0) red[tid >> 5] = ss;
  __syncthreads();
  float tot = 0.f;
#pragma unroll
  for (int w8 = 0; w8 < 8; ++w8) tot += red[w8];
  float rr = rsqrtf(tot / (float)D_MODEL + EPSV);
#pragma unroll
  for (int j = 0; j < 4; ++j)
    zr[tid + j*256] = f2bf(vals[j] * rr * w[tid + j*256]);
}

// Generic GEMM: C[M,N] = A[M,K](bf16,rm) @ B[K,N] with B given TRANSPOSED as
// Bt[N][K] (bf16,rm). One wave -> 32x64 tile (2 A-frags, 8 accums, all-b128 loads).
// outF!=null: f32 out (+ optional residual). else bf16 out.
__global__ __launch_bounds__(32)
void gemm_bf16_kernel(const u16t* __restrict__ A, const u16t* __restrict__ Bt,
                      int M, int N, int K,
                      const float* __restrict__ resid,
                      float* __restrict__ outF, u16t* __restrict__ outB) {
  int n0 = blockIdx.x * 64;
  int r0 = blockIdx.y * 32;
  const v8f zr = {0.f,0.f,0.f,0.f,0.f,0.f,0.f,0.f};
  v8f acc[2][4] = {{zr, zr, zr, zr}, {zr, zr, zr, zr}};
  for (int k0 = 0; k0 < K; k0 += 32) {
    if (k0 + 64 < K)
      __builtin_prefetch(A + (size_t)(r0 + (threadIdx.x & 31)) * K + k0 + 64, 0, 1);
    v16bf a0 = load_a_rm(A, K, r0,      k0);
    v16bf a1 = load_a_rm(A, K, r0 + 16, k0);
#pragma unroll
    for (int i = 0; i < 4; ++i) {
      v16bf b = load_b_tr(Bt, K, k0, n0 + i * 16);
      acc[0][i] = wmma_bf16(a0, b, acc[0][i]);
      acc[1][i] = wmma_bf16(a1, b, acc[1][i]);
    }
  }
  int l = threadIdx.x & 31, half = l >> 4, c = l & 15;
#pragma unroll
  for (int mi = 0; mi < 2; ++mi)
#pragma unroll
    for (int i = 0; i < 4; ++i)
#pragma unroll
      for (int r = 0; r < 8; ++r) {
        size_t idx = (size_t)(r0 + mi * 16 + r + half * 8) * N + n0 + i * 16 + c;
        float v = acc[mi][i][r];
        if (outF) { if (resid) v += resid[idx]; outF[idx] = v; }
        else        outB[idx] = f2bf(v);
      }
}

// Fused SwiGLU: out = silu(A@B1) * (A@B2), Bt-transposed weights, bf16 out.
// One wave -> 32x64 tile, 16 accumulators (2 A-frags shared across both gates).
__global__ __launch_bounds__(32)
void gemm_swiglu_kernel(const u16t* __restrict__ A, const u16t* __restrict__ B1t,
                        const u16t* __restrict__ B2t, int M, int N, int K,
                        u16t* __restrict__ out) {
  int n0 = blockIdx.x * 64;
  int r0 = blockIdx.y * 32;
  const v8f zr = {0.f,0.f,0.f,0.f,0.f,0.f,0.f,0.f};
  v8f u[2][4] = {{zr, zr, zr, zr}, {zr, zr, zr, zr}};
  v8f w[2][4] = {{zr, zr, zr, zr}, {zr, zr, zr, zr}};
  for (int k0 = 0; k0 < K; k0 += 32) {
    v16bf a0 = load_a_rm(A, K, r0,      k0);
    v16bf a1 = load_a_rm(A, K, r0 + 16, k0);
#pragma unroll
    for (int i = 0; i < 4; ++i) {
      v16bf b = load_b_tr(B1t, K, k0, n0 + i * 16);
      u[0][i] = wmma_bf16(a0, b, u[0][i]);
      u[1][i] = wmma_bf16(a1, b, u[1][i]);
    }
#pragma unroll
    for (int i = 0; i < 4; ++i) {
      v16bf b = load_b_tr(B2t, K, k0, n0 + i * 16);
      w[0][i] = wmma_bf16(a0, b, w[0][i]);
      w[1][i] = wmma_bf16(a1, b, w[1][i]);
    }
  }
  int l = threadIdx.x & 31, half = l >> 4, c = l & 15;
#pragma unroll
  for (int mi = 0; mi < 2; ++mi)
#pragma unroll
    for (int i = 0; i < 4; ++i)
#pragma unroll
      for (int r = 0; r < 8; ++r) {
        float uu = u[mi][i][r];
        float g  = (uu / (1.f + __expf(-uu))) * w[mi][i][r];
        out[(size_t)(r0 + mi * 16 + r + half * 8) * N + n0 + i * 16 + c] = f2bf(g);
      }
}

// Flash-style causal GQA attention. One wave per (b, head, 16-query tile).
// q: [b][TP][H*DK], k: [b][TP][KVW] (token-major => contiguous for Q@K^T),
// vt: [b][KVW][TPAD] (feature-major => contiguous for P@V). ctx out:
// [b][SEQ][H*DV] bf16, gw rows dropped, headwise RMSNorm applied.
__global__ __launch_bounds__(32)
void attn_kernel(const u16t* __restrict__ q, const u16t* __restrict__ kM,
                 const u16t* __restrict__ vt, const float* __restrict__ hn_w,
                 u16t* __restrict__ ctx) {
  int qt   = blockIdx.x;
  int head = blockIdx.y;
  int b    = blockIdx.z;
  int g    = head >> 2;   // N_HEADS / GROUP_KV = 4
  int l = threadIdx.x & 31, half = l >> 4, lc = l & 15;

  const u16t* qbase  = q  + (size_t)b * TP * (N_HEADS * DK) + head * DK;
  const u16t* kbase  = kM + (size_t)b * TP * KVW + g * DK;
  const u16t* vtbase = vt + (size_t)(b * KVW + g * DV) * TPAD;

  v16bf qa0 = load_a_rm(qbase, N_HEADS * DK, qt * 16, 0);
  v16bf qa1 = load_a_rm(qbase, N_HEADS * DK, qt * 16, 32);

  const v8f zrv = {0.f,0.f,0.f,0.f,0.f,0.f,0.f,0.f};
  v8f o[4] = {zrv, zrv, zrv, zrv};
  float rm[8], rs[8];
#pragma unroll
  for (int r = 0; r < 8; ++r) { rm[r] = -1e30f; rs[r] = 0.f; }

  __shared__ u16t pls[16 * 32];   // P tile staging (C-layout -> A-layout transpose)

  int qmax = qt * 16 + 15;
  for (int j0 = 0; j0 <= qmax; j0 += 32) {
    // S = Q @ K^T  (16 x 32 keys, two f32 accumulators); K loads contiguous
    v8f s0 = zrv, s1 = zrv;
    s0 = wmma_bf16(qa0, load_b_tr(kbase, KVW, 0,  j0),      s0);
    s0 = wmma_bf16(qa1, load_b_tr(kbase, KVW, 32, j0),      s0);
    s1 = wmma_bf16(qa0, load_b_tr(kbase, KVW, 0,  j0 + 16), s1);
    s1 = wmma_bf16(qa1, load_b_tr(kbase, KVW, 32, j0 + 16), s1);

    // online softmax (rows r+8*half live across the 16 lanes of this half)
#pragma unroll
    for (int r = 0; r < 8; ++r) {
      int qrow = qt * 16 + r + half * 8;
      float e0 = s0[r] * QK_SCALE;
      float e1 = s1[r] * QK_SCALE;
      if (j0 + lc > qrow)       e0 = -1e30f;
      if (j0 + 16 + lc > qrow)  e1 = -1e30f;
      float mx = fmaxf(e0, e1);
      for (int m = 8; m >= 1; m >>= 1) mx = fmaxf(mx, __shfl_xor(mx, m, 32));
      float mnew = fmaxf(rm[r], mx);
      float p0 = __expf(e0 - mnew);
      float p1 = __expf(e1 - mnew);
      float ps = p0 + p1;
      for (int m = 8; m >= 1; m >>= 1) ps += __shfl_xor(ps, m, 32);
      float corr = __expf(rm[r] - mnew);
      rs[r] = rs[r] * corr + ps;
      rm[r] = mnew;
#pragma unroll
      for (int i = 0; i < 4; ++i) o[i][r] = o[i][r] * corr;
      int prow = r + half * 8;
      pls[prow * 32 + lc]      = f2bf(p0);
      pls[prow * 32 + 16 + lc] = f2bf(p1);
    }
    __asm__ volatile("s_wait_dscnt 0x0" ::: "memory");

    // P (16x32) as A-frag from LDS: contiguous per lane -> two ds_load_b128
    FragBF pa;
    {
      const u16t* pp = &pls[lc * 32 + half * 16];
      pa.q[0] = *(const uint4*)(pp);        // K 0-7  (or 16-23)
      pa.q[1] = *(const uint4*)(pp + 8);    // K 8-15 (or 24-31)
    }
    // O += P @ V : B[k][n] = Vt[g*DV+n][j0+k] -> contiguous b128 loads
#pragma unroll
    for (int i = 0; i < 4; ++i) {
      v16bf vb = load_b_tr(vtbase, TPAD, j0, i * 16);
      o[i] = wmma_bf16(pa.v, vb, o[i]);
    }
  }

  // epilogue: 1/rowsum, headwise RMSNorm over dv=64, per-head weight, store bf16
#pragma unroll
  for (int r = 0; r < 8; ++r) {
    float inv = 1.f / rs[r];
    float ssq = 0.f;
#pragma unroll
    for (int i = 0; i < 4; ++i) { float vv = o[i][r] * inv; o[i][r] = vv; ssq += vv * vv; }
    for (int m = 8; m >= 1; m >>= 1) ssq += __shfl_xor(ssq, m, 32);
    float rr = rsqrtf(ssq / (float)DV + EPSV);
    int qrow = qt * 16 + r + half * 8;
    int s = qrow - GWTOK;
    if (s >= 0 && s < SEQ) {
      u16t* orow = ctx + ((size_t)b * SEQ + s) * (N_HEADS * DV) + head * DV;
#pragma unroll
      for (int i = 0; i < 4; ++i)
        orow[i * 16 + lc] = f2bf(o[i][r] * rr * hn_w[head * DV + i * 16 + lc]);
    }
  }
}

// ---------------- host launcher ----------------
extern "C" void kernel_launch(void* const* d_in, const int* in_sizes, int n_in,
                              void* d_out, int out_size, void* d_ws, size_t ws_size,
                              hipStream_t stream) {
  const float* x      = (const float*)d_in[0];
  const float* nin_w  = (const float*)d_in[1];
  const float* gw     = (const float*)d_in[2];
  const float* Wq     = (const float*)d_in[3];
  const float* Wk     = (const float*)d_in[4];
  const float* Wv     = (const float*)d_in[5];
  const float* Wo     = (const float*)d_in[6];
  const float* hn_w   = (const float*)d_in[7];
  const float* ff_w   = (const float*)d_in[8];
  const float* W1     = (const float*)d_in[9];
  const float* W3     = (const float*)d_in[10];
  const float* W2     = (const float*)d_in[11];
  float* out = (float*)d_out;

  char* p = (char*)d_ws;
  auto carve = [&](size_t bytes) -> void* {
    void* r = (void*)p;
    p += (bytes + 255) & ~(size_t)255;
    return r;
  };
  u16t* hB   = (u16t*)carve((size_t)MQ * D_MODEL * 2);
  u16t* qB   = (u16t*)carve((size_t)MQ * (N_HEADS*DK) * 2);
  u16t* kB   = (u16t*)carve((size_t)(MQ + 32) * KVW * 2);  // +tail slack for chunked K reads
  u16t* vB   = (u16t*)carve((size_t)MQ * KVW * 2);
  u16t* vtB  = (u16t*)carve((size_t)BATCH * KVW * TPAD * 2);
  u16t* ctxB = (u16t*)carve((size_t)MY * (N_HEADS*DV) * 2);
  float* yF  = (float*)carve((size_t)MY * D_MODEL * 4);
  u16t* zB   = (u16t*)carve((size_t)MY * D_MODEL * 2);
  u16t* gB   = (u16t*)carve((size_t)MY * D_FF * 2);
  u16t* WqT  = (u16t*)carve((size_t)D_MODEL * (N_HEADS*DK) * 2);
  u16t* WkT  = (u16t*)carve((size_t)D_MODEL * KVW * 2);
  u16t* WvT  = (u16t*)carve((size_t)D_MODEL * KVW * 2);
  u16t* WoT  = (u16t*)carve((size_t)(N_HEADS*DV) * D_MODEL * 2);
  u16t* W1T  = (u16t*)carve((size_t)D_MODEL * D_FF * 2);
  u16t* W3T  = (u16t*)carve((size_t)D_MODEL * D_FF * 2);
  u16t* W2T  = (u16t*)carve((size_t)D_FF * D_MODEL * 2);

  // 1) cast + transpose weights fp32 [K][N] -> bf16 [N][K] (LDS-tiled, coalesced)
  struct { const float* src; u16t* dst; int K, N; } casts[7] = {
    { Wq, WqT, D_MODEL, N_HEADS * DK },
    { Wk, WkT, D_MODEL, KVW },
    { Wv, WvT, D_MODEL, KVW },
    { Wo, WoT, N_HEADS * DV, D_MODEL },
    { W1, W1T, D_MODEL, D_FF },
    { W3, W3T, D_MODEL, D_FF },
    { W2, W2T, D_FF, D_MODEL },
  };
  for (int i = 0; i < 7; ++i)
    cast_transpose_kernel<<<dim3(casts[i].N / 32, casts[i].K / 32), 256, 0, stream>>>(
        casts[i].src, casts[i].dst, casts[i].K, casts[i].N);

  // 2) h = [gw ; rms(x)] padded to TP rows, bf16
  prep_h_kernel<<<dim3(TP, BATCH), 256, 0, stream>>>(x, nin_w, gw, hB);

  // 3) Q/K/V projections (WMMA bf16, 32x64 tiles, all-b128 operand loads)
  gemm_bf16_kernel<<<dim3((N_HEADS*DK)/64, MQ/32), 32, 0, stream>>>(
      hB, WqT, MQ, N_HEADS*DK, D_MODEL, nullptr, nullptr, qB);
  gemm_bf16_kernel<<<dim3(KVW/64, MQ/32), 32, 0, stream>>>(
      hB, WkT, MQ, KVW, D_MODEL, nullptr, nullptr, kB);
  gemm_bf16_kernel<<<dim3(KVW/64, MQ/32), 32, 0, stream>>>(
      hB, WvT, MQ, KVW, D_MODEL, nullptr, nullptr, vB);

  // 3b) V -> feature-major Vt for contiguous P@V fragment loads (tiled transpose)
  transpose_v_kernel<<<dim3((TPAD + 31) / 32, KVW / 32, BATCH), 256, 0, stream>>>(vB, vtB);

  // 4) causal GQA flash attention + headwise RMSNorm -> ctx (gw rows dropped)
  attn_kernel<<<dim3(TP/16, N_HEADS, BATCH), 32, 0, stream>>>(qB, kB, vtB, hn_w, ctxB);

  // 5) y = x + ctx @ Wo  (f32 out)
  gemm_bf16_kernel<<<dim3(D_MODEL/64, MY/32), 32, 0, stream>>>(
      ctxB, WoT, MY, D_MODEL, N_HEADS*DV, x, yF, nullptr);

  // 6) z = rms(y) * ff_norm_w  (bf16)
  rms_rows_kernel<<<MY, 256, 0, stream>>>(yF, ff_w, zB);

  // 7) g = silu(z@W1) * (z@W3)  (fused, 32x64 tiles, bf16)
  gemm_swiglu_kernel<<<dim3(D_FF/64, MY/32), 32, 0, stream>>>(
      zB, W1T, W3T, MY, D_FF, D_MODEL, gB);

  // 8) out = y + g @ W2  (f32, final output)
  gemm_bf16_kernel<<<dim3(D_MODEL/64, MY/32), 32, 0, stream>>>(
      gB, W2T, MY, D_MODEL, D_FF, yF, out, nullptr);
}
